// LanguageModel_29059748725260
// MI455X (gfx1250) — matrix operand order repaired and checked by reading, not compiled, not run on hardware
//
#include <hip/hip_runtime.h>

// ---------------------------------------------------------------------------
// Model dims
// ---------------------------------------------------------------------------
#define LAYERS 12
#define HEADS  12
#define DMODEL 768
#define FDIM   3072
#define VOCAB  21128
#define VPAD   21184          // round up to multiple of 64
#define SEQ    320
#define BATCH  16
#define DHEAD  64
#define TOKENS (BATCH*SEQ)    // 5120
#define TITLE  20
#define TSTART 300
#define MCLS   (BATCH*TITLE)  // 320

typedef __attribute__((ext_vector_type(16))) __bf16 v16bf;
typedef __attribute__((ext_vector_type(8)))  float  v8f;

union Frag {
  v16bf v;
  uint4 q[2];
  unsigned short s[16];
};

__device__ __forceinline__ unsigned short f2bf(float f) {
  union { float f; unsigned u; } x; x.f = f;
  unsigned r = x.u + 0x7FFFu + ((x.u >> 16) & 1u);   // round-to-nearest-even
  return (unsigned short)(r >> 16);
}

__device__ __forceinline__ v8f wmma_bf16(const v16bf& a, const v16bf& b, const v8f& c) {
  return __builtin_amdgcn_wmma_f32_16x16x32_bf16(false, a, false, b, (short)0, c, false, false);
}

// ---------------------------------------------------------------------------
// Weight convert + transpose: src[K][N] f32 -> dst[Npad][K] bf16 (zero pad)
// ---------------------------------------------------------------------------
__global__ void cvt_t_kernel(const float* __restrict__ src, unsigned short* __restrict__ dst,
                             int K, int N, int Npad) {
  long idx = (long)blockIdx.x * blockDim.x + threadIdx.x;
  long total = (long)Npad * K;
  if (idx >= total) return;
  int n = (int)(idx / K);
  int k = (int)(idx % K);
  dst[idx] = (n < N) ? f2bf(src[(long)k * N + n]) : (unsigned short)0;
}

// ---------------------------------------------------------------------------
// Embedding + LayerNorm : one block per token, 256 threads over D=768
// ---------------------------------------------------------------------------
__global__ void embed_ln_kernel(const int* __restrict__ x, const float* __restrict__ we,
                                const float* __restrict__ pe, const float* __restrict__ te,
                                const float* __restrict__ g, const float* __restrict__ bta,
                                float* __restrict__ h, unsigned short* __restrict__ hb) {
  int tok = blockIdx.x;
  int s = tok % SEQ;
  int id = x[tok];
  const float* wrow = we + (long)id * DMODEL;
  const float* prow = pe + (long)s * DMODEL;
  __shared__ float red[256];
  float loc[3];
  float sum = 0.f;
#pragma unroll
  for (int i = 0; i < 3; ++i) {
    int c = threadIdx.x + i * 256;
    float v = wrow[c] + prow[c] + te[c];
    loc[i] = v; sum += v;
  }
  red[threadIdx.x] = sum; __syncthreads();
  for (int st = 128; st > 0; st >>= 1) {
    if (threadIdx.x < st) red[threadIdx.x] += red[threadIdx.x + st];
    __syncthreads();
  }
  float mean = red[0] * (1.f / DMODEL);
  __syncthreads();
  float vs = 0.f;
#pragma unroll
  for (int i = 0; i < 3; ++i) { float d = loc[i] - mean; vs += d * d; }
  red[threadIdx.x] = vs; __syncthreads();
  for (int st = 128; st > 0; st >>= 1) {
    if (threadIdx.x < st) red[threadIdx.x] += red[threadIdx.x + st];
    __syncthreads();
  }
  float rstd = rsqrtf(red[0] * (1.f / DMODEL) + 1e-12f);
#pragma unroll
  for (int i = 0; i < 3; ++i) {
    int c = threadIdx.x + i * 256;
    float nv = (loc[i] - mean) * rstd * g[c] + bta[c];
    h[(long)tok * DMODEL + c]  = nv;
    hb[(long)tok * DMODEL + c] = f2bf(nv);
  }
}

// ---------------------------------------------------------------------------
// Residual + LayerNorm : h = LN(h + delta), writes f32 + bf16 copies
// ---------------------------------------------------------------------------
__global__ void residual_ln_kernel(float* __restrict__ h, unsigned short* __restrict__ hb,
                                   const float* __restrict__ delta,
                                   const float* __restrict__ g, const float* __restrict__ bta) {
  int tok = blockIdx.x;
  __shared__ float red[256];
  float loc[3];
  float sum = 0.f;
#pragma unroll
  for (int i = 0; i < 3; ++i) {
    int c = threadIdx.x + i * 256;
    float v = h[(long)tok * DMODEL + c] + delta[(long)tok * DMODEL + c];
    loc[i] = v; sum += v;
  }
  red[threadIdx.x] = sum; __syncthreads();
  for (int st = 128; st > 0; st >>= 1) {
    if (threadIdx.x < st) red[threadIdx.x] += red[threadIdx.x + st];
    __syncthreads();
  }
  float mean = red[0] * (1.f / DMODEL);
  __syncthreads();
  float vs = 0.f;
#pragma unroll
  for (int i = 0; i < 3; ++i) { float d = loc[i] - mean; vs += d * d; }
  red[threadIdx.x] = vs; __syncthreads();
  for (int st = 128; st > 0; st >>= 1) {
    if (threadIdx.x < st) red[threadIdx.x] += red[threadIdx.x + st];
    __syncthreads();
  }
  float rstd = rsqrtf(red[0] * (1.f / DMODEL) + 1e-12f);
#pragma unroll
  for (int i = 0; i < 3; ++i) {
    int c = threadIdx.x + i * 256;
    float nv = (loc[i] - mean) * rstd * g[c] + bta[c];
    h[(long)tok * DMODEL + c]  = nv;
    hb[(long)tok * DMODEL + c] = f2bf(nv);
  }
}

// ---------------------------------------------------------------------------
// WMMA GEMM: C[M x N] = A_bf16[M x K] @ Bt_bf16[N x K]^T + bias
// 8 waves/block; each wave owns a 32(M) x 64(N) strip (2x4 WMMA tiles),
// reusing each B-fragment across two A-fragments: 12 b128 loads per
// 8 WMMAs (vs 10 per 4 for a 16x64 strip) -> 40% less VMEM issue per op.
// MODE 0: f32 out (row-major, ld = N, cols >= Nreal -> 0)
// MODE 1: bf16 out with exact GELU
// MODE 2: bf16 scatter to [B,H,S,DH]   (q / k)
// MODE 3: bf16 scatter to [B,H,DH,S]   (v transposed)
// ---------------------------------------------------------------------------
template <int MODE>
__global__ void gemm_wmma(const unsigned short* __restrict__ A,
                          const unsigned short* __restrict__ Bt,
                          const float* __restrict__ bias,
                          void* __restrict__ Cout,
                          int M, int N, int K, int Nreal) {
  const int wid = (int)((blockIdx.x * (long)blockDim.x + threadIdx.x) >> 5);
  const int l  = threadIdx.x & 31;
  const int hi = l >> 4;
  const int ln = l & 15;
  const int n4 = N >> 6;
  const int mT = wid / n4;              // 32-row strip index
  const int nT = wid - mT * n4;
  if (mT * 32 >= M) return;
  const int row0 = mT * 32;
  const int col0 = nT * 64;

  v8f acc[2][4] = {{v8f{}, v8f{}, v8f{}, v8f{}}, {v8f{}, v8f{}, v8f{}, v8f{}}};
  const unsigned short* aBase0 = A + (long)(row0 + ln) * K + hi * 16;
  const unsigned short* aBase1 = aBase0 + (long)16 * K;
  const unsigned short* bBase  = Bt + (long)(col0 + ln) * K + hi * 16;

  for (int kt = 0; kt < K; kt += 32) {
    Frag a0, a1;
    a0.q[0] = *(const uint4*)(aBase0 + kt);
    a0.q[1] = *(const uint4*)(aBase0 + kt + 8);
    a1.q[0] = *(const uint4*)(aBase1 + kt);
    a1.q[1] = *(const uint4*)(aBase1 + kt + 8);
#pragma unroll
    for (int j = 0; j < 4; ++j) {
      v16bf bv = *(const v16bf*)(bBase + (long)j * 16 * K + kt);
      acc[0][j] = wmma_bf16(a0.v, bv, acc[0][j]);
      acc[1][j] = wmma_bf16(a1.v, bv, acc[1][j]);
    }
  }

#pragma unroll
  for (int mi = 0; mi < 2; ++mi) {
#pragma unroll
    for (int j = 0; j < 4; ++j) {
#pragma unroll
      for (int i = 0; i < 8; ++i) {
        int r = row0 + mi * 16 + hi * 8 + i;
        int c = col0 + j * 16 + ln;
        float v = acc[mi][j][i];
        if (MODE == 0) {
          float o = (c < Nreal) ? (v + bias[c]) : 0.f;
          ((float*)Cout)[(long)r * N + c] = o;
        } else if (MODE == 1) {
          float xg = v + bias[c];
          float o = 0.5f * xg * (1.f + erff(xg * 0.70710678118654752f));
          ((unsigned short*)Cout)[(long)r * N + c] = f2bf(o);
        } else if (MODE == 2) {
          int b = r / SEQ, s = r - b * SEQ;
          int hh = c >> 6, dh = c & 63;
          float o = v + bias[c];
          ((unsigned short*)Cout)[(((long)(b * HEADS + hh) * SEQ + s) << 6) + dh] = f2bf(o);
        } else {
          int b = r / SEQ, s = r - b * SEQ;
          int hh = c >> 6, dh = c & 63;
          float o = v + bias[c];
          ((unsigned short*)Cout)[(long)((b * HEADS + hh) * DHEAD + dh) * SEQ + s] = f2bf(o);
        }
      }
    }
  }
}

// ---------------------------------------------------------------------------
// Fused attention: one wave per (b,h, 16-query tile).
// scores = (q k^T)/8 + mask  -> softmax(row) -> ctx = P @ V
// q,k : [B,H,S,DH] bf16 ; vT : [B,H,DH,S] bf16 ; ctx : [B*S, D] bf16
// ---------------------------------------------------------------------------
__global__ void attn_kernel(const unsigned short* __restrict__ q,
                            const unsigned short* __restrict__ k,
                            const unsigned short* __restrict__ vT,
                            unsigned short* __restrict__ ctx) {
  __shared__ float sc[16][SEQ];            // 20 KB
  __shared__ unsigned short pb[16][SEQ];   // 10 KB
  const int qt = blockIdx.x;               // 0..19
  const int bh = blockIdx.y;               // 0..191
  const int b = bh / HEADS, h = bh - b * HEADS;
  const int l = threadIdx.x & 31;
  const int hi = l >> 4, ln = l & 15;

  // q A-fragments for K = 0..31 and 32..63
  const unsigned short* qBase = q + ((long)bh * SEQ + qt * 16 + ln) * DHEAD + hi * 16;
  Frag qa0, qa1;
  qa0.q[0] = *(const uint4*)(qBase);
  qa0.q[1] = *(const uint4*)(qBase + 8);
  qa1.q[0] = *(const uint4*)(qBase + 32);
  qa1.q[1] = *(const uint4*)(qBase + 40);

  for (int jt = 0; jt < SEQ / 16; ++jt) {
    const unsigned short* kBase = k + ((long)bh * SEQ + jt * 16 + ln) * DHEAD + hi * 16;
    v16bf kb0 = *(const v16bf*)(kBase);
    v16bf kb1 = *(const v16bf*)(kBase + 32);
    v8f a = {};
    a = wmma_bf16(qa0.v, kb0, a);
    a = wmma_bf16(qa1.v, kb1, a);
#pragma unroll
    for (int i = 0; i < 8; ++i) {
      int rloc = hi * 8 + i;
      int row = qt * 16 + rloc;
      int col = jt * 16 + ln;
      float s = a[i] * 0.125f;                    // 1/sqrt(64)
      if (col >= TSTART && col > row) s -= 10000.f;
      sc[rloc][col] = s;
    }
  }
  __syncthreads();

  if (l < 16) {
    float m = -3.4e38f;
    for (int j = 0; j < SEQ; ++j) m = fmaxf(m, sc[l][j]);
    float sum = 0.f;
    for (int j = 0; j < SEQ; ++j) { float e = __expf(sc[l][j] - m); sc[l][j] = e; sum += e; }
    float inv = 1.f / sum;
    for (int j = 0; j < SEQ; ++j) pb[l][j] = f2bf(sc[l][j] * inv);
  }
  __syncthreads();

  v8f acc[4] = {v8f{}, v8f{}, v8f{}, v8f{}};
  const unsigned short* vBase = vT + ((long)bh * DHEAD + ln) * SEQ + hi * 16;
  for (int kt = 0; kt < SEQ; kt += 32) {
    Frag pa;
    const unsigned short* pRow = &pb[ln][0] + kt + hi * 16;
    pa.q[0] = *(const uint4*)(pRow);
    pa.q[1] = *(const uint4*)(pRow + 8);
#pragma unroll
    for (int j = 0; j < 4; ++j) {
      v16bf bv = *(const v16bf*)(vBase + (long)j * 16 * SEQ + kt);
      acc[j] = wmma_bf16(pa.v, bv, acc[j]);
    }
  }

#pragma unroll
  for (int j = 0; j < 4; ++j) {
#pragma unroll
    for (int i = 0; i < 8; ++i) {
      int rloc = hi * 8 + i;
      long tok = (long)b * SEQ + qt * 16 + rloc;
      int c = h * DHEAD + j * 16 + ln;
      ctx[tok * DMODEL + c] = f2bf(acc[j][i]);
    }
  }
}

// ---------------------------------------------------------------------------
// Gather last TITLE tokens' bf16 activations for the classifier
// ---------------------------------------------------------------------------
__global__ void gather_cls_kernel(const unsigned short* __restrict__ hb,
                                  unsigned short* __restrict__ a) {
  long idx = (long)blockIdx.x * blockDim.x + threadIdx.x;
  if (idx >= (long)MCLS * DMODEL) return;
  int r = (int)(idx / DMODEL), c = (int)(idx % DMODEL);
  int b = r / TITLE, t = r - b * TITLE;
  a[idx] = hb[((long)b * SEQ + TSTART + t) * DMODEL + c];
}

// ---------------------------------------------------------------------------
// Per-row log-softmax NLL, then deterministic mean
// ---------------------------------------------------------------------------
__global__ void nll_kernel(const float* __restrict__ logits, const int* __restrict__ x,
                           float* __restrict__ nll) {
  int r = blockIdx.x;                       // 0..319
  int b = r / TITLE, t = r - b * TITLE;
  int lbl = x[b * SEQ + TSTART + t];
  const float* row = logits + (long)r * VPAD;
  __shared__ float red[256];
  float m = -3.4e38f;
  for (int j = threadIdx.x; j < VOCAB; j += 256) m = fmaxf(m, row[j]);
  red[threadIdx.x] = m; __syncthreads();
  for (int st = 128; st > 0; st >>= 1) {
    if (threadIdx.x < st) red[threadIdx.x] = fmaxf(red[threadIdx.x], red[threadIdx.x + st]);
    __syncthreads();
  }
  m = red[0]; __syncthreads();
  float sum = 0.f;
  for (int j = threadIdx.x; j < VOCAB; j += 256) sum += __expf(row[j] - m);
  red[threadIdx.x] = sum; __syncthreads();
  for (int st = 128; st > 0; st >>= 1) {
    if (threadIdx.x < st) red[threadIdx.x] += red[threadIdx.x + st];
    __syncthreads();
  }
  if (threadIdx.x == 0) nll[r] = -(row[lbl] - m - logf(red[0]));
}

__global__ void mean_kernel(const float* __restrict__ nll, float* __restrict__ out) {
  if (threadIdx.x == 0 && blockIdx.x == 0) {
    float s = 0.f;
    for (int i = 0; i < MCLS; ++i) s += nll[i];
    out[0] = s / (float)MCLS;
  }
}

// ---------------------------------------------------------------------------
// Host launcher
// ---------------------------------------------------------------------------
extern "C" void kernel_launch(void* const* d_in, const int* in_sizes, int n_in,
                              void* d_out, int out_size, void* d_ws, size_t ws_size,
                              hipStream_t stream) {
  (void)in_sizes; (void)n_in; (void)out_size; (void)ws_size;
  const float* word_emb = (const float*)d_in[0];
  const float* pos_emb  = (const float*)d_in[1];
  const float* type_emb = (const float*)d_in[2];
  const float* ln_e_g   = (const float*)d_in[3];
  const float* ln_e_b   = (const float*)d_in[4];
  const float* Wq   = (const float*)d_in[5];
  const float* bq   = (const float*)d_in[6];
  const float* Wk   = (const float*)d_in[7];
  const float* bk   = (const float*)d_in[8];
  const float* Wv   = (const float*)d_in[9];
  const float* bv   = (const float*)d_in[10];
  const float* Wo   = (const float*)d_in[11];
  const float* bo   = (const float*)d_in[12];
  const float* ln1g = (const float*)d_in[13];
  const float* ln1b = (const float*)d_in[14];
  const float* Wi   = (const float*)d_in[15];
  const float* bi   = (const float*)d_in[16];
  const float* Wf   = (const float*)d_in[17];
  const float* bf_  = (const float*)d_in[18];
  const float* ln2g = (const float*)d_in[19];
  const float* ln2b = (const float*)d_in[20];
  const float* Wcls = (const float*)d_in[21];
  const float* bcls = (const float*)d_in[22];
  const int*   x    = (const int*)d_in[23];
  float* out = (float*)d_out;

  // ---- workspace carve-out (256B aligned) ----
  char* p = (char*)d_ws;
  auto alloc = [&](size_t bytes) -> char* {
    char* r = p; p += (bytes + 255) & ~size_t(255); return r;
  };
  unsigned short* WqT = (unsigned short*)alloc((size_t)LAYERS * DMODEL * DMODEL * 2);
  unsigned short* WkT = (unsigned short*)alloc((size_t)LAYERS * DMODEL * DMODEL * 2);
  unsigned short* WvT = (unsigned short*)alloc((size_t)LAYERS * DMODEL * DMODEL * 2);
  unsigned short* WoT = (unsigned short*)alloc((size_t)LAYERS * DMODEL * DMODEL * 2);
  unsigned short* WiT = (unsigned short*)alloc((size_t)LAYERS * FDIM * DMODEL * 2);   // [F][D]
  unsigned short* WfT = (unsigned short*)alloc((size_t)LAYERS * DMODEL * FDIM * 2);   // [D][F]
  unsigned short* WcT = (unsigned short*)alloc((size_t)VPAD * DMODEL * 2);
  float*          h    = (float*)alloc((size_t)TOKENS * DMODEL * 4);
  unsigned short* hb   = (unsigned short*)alloc((size_t)TOKENS * DMODEL * 2);
  unsigned short* qb   = (unsigned short*)alloc((size_t)TOKENS * DMODEL * 2);
  unsigned short* kb   = (unsigned short*)alloc((size_t)TOKENS * DMODEL * 2);
  unsigned short* vTb  = (unsigned short*)alloc((size_t)TOKENS * DMODEL * 2);
  unsigned short* ctxb = (unsigned short*)alloc((size_t)TOKENS * DMODEL * 2);
  float*          tmpf = (float*)alloc((size_t)TOKENS * DMODEL * 4);
  unsigned short* ffib = (unsigned short*)alloc((size_t)TOKENS * FDIM * 2);
  unsigned short* a320 = (unsigned short*)alloc((size_t)MCLS * DMODEL * 2);
  float*          logits = (float*)alloc((size_t)MCLS * VPAD * 4);
  float*          nllws  = (float*)alloc((size_t)MCLS * 4);

  auto cvt = [&](const float* src, unsigned short* dst, int K, int N, int Npad) {
    long cnt = (long)Npad * K;
    cvt_t_kernel<<<dim3((unsigned)((cnt + 255) / 256)), 256, 0, stream>>>(src, dst, K, N, Npad);
  };
  // weight packing (bf16, transposed to [N][K])
  for (int l = 0; l < LAYERS; ++l) {
    cvt(Wq + (long)l * DMODEL * DMODEL, WqT + (long)l * DMODEL * DMODEL, DMODEL, DMODEL, DMODEL);
    cvt(Wk + (long)l * DMODEL * DMODEL, WkT + (long)l * DMODEL * DMODEL, DMODEL, DMODEL, DMODEL);
    cvt(Wv + (long)l * DMODEL * DMODEL, WvT + (long)l * DMODEL * DMODEL, DMODEL, DMODEL, DMODEL);
    cvt(Wo + (long)l * DMODEL * DMODEL, WoT + (long)l * DMODEL * DMODEL, DMODEL, DMODEL, DMODEL);
    cvt(Wi + (long)l * DMODEL * FDIM,   WiT + (long)l * FDIM * DMODEL,   DMODEL, FDIM,   FDIM);
    cvt(Wf + (long)l * FDIM * DMODEL,   WfT + (long)l * DMODEL * FDIM,   FDIM,   DMODEL, DMODEL);
  }
  cvt(Wcls, WcT, DMODEL, VOCAB, VPAD);

  // embeddings + LN
  embed_ln_kernel<<<TOKENS, 256, 0, stream>>>(x, word_emb, pos_emb, type_emb, ln_e_g, ln_e_b, h, hb);

  auto gemmGrid = [](int M, int N) -> dim3 {
    long waves = (long)(M / 32) * (N / 64);
    return dim3((unsigned)((waves + 7) / 8));
  };

  for (int l = 0; l < LAYERS; ++l) {
    const unsigned short* wq = WqT + (long)l * DMODEL * DMODEL;
    const unsigned short* wk = WkT + (long)l * DMODEL * DMODEL;
    const unsigned short* wv = WvT + (long)l * DMODEL * DMODEL;
    const unsigned short* wo = WoT + (long)l * DMODEL * DMODEL;
    const unsigned short* wi = WiT + (long)l * FDIM * DMODEL;
    const unsigned short* wf = WfT + (long)l * DMODEL * FDIM;

    // q, k -> [B,H,S,DH]; v -> [B,H,DH,S]
    gemm_wmma<2><<<gemmGrid(TOKENS, DMODEL), 256, 0, stream>>>(
        hb, wq, bq + (long)l * DMODEL, qb, TOKENS, DMODEL, DMODEL, DMODEL);
    gemm_wmma<2><<<gemmGrid(TOKENS, DMODEL), 256, 0, stream>>>(
        hb, wk, bk + (long)l * DMODEL, kb, TOKENS, DMODEL, DMODEL, DMODEL);
    gemm_wmma<3><<<gemmGrid(TOKENS, DMODEL), 256, 0, stream>>>(
        hb, wv, bv + (long)l * DMODEL, vTb, TOKENS, DMODEL, DMODEL, DMODEL);

    // fused attention
    attn_kernel<<<dim3(SEQ / 16, BATCH * HEADS), 32, 0, stream>>>(qb, kb, vTb, ctxb);

    // output projection + residual LN
    gemm_wmma<0><<<gemmGrid(TOKENS, DMODEL), 256, 0, stream>>>(
        ctxb, wo, bo + (long)l * DMODEL, tmpf, TOKENS, DMODEL, DMODEL, DMODEL);
    residual_ln_kernel<<<TOKENS, 256, 0, stream>>>(
        h, hb, tmpf, ln1g + (long)l * DMODEL, ln1b + (long)l * DMODEL);

    // FF1 (GELU, bf16 out) and FF2 (f32 out) + residual LN
    gemm_wmma<1><<<gemmGrid(TOKENS, FDIM), 256, 0, stream>>>(
        hb, wi, bi + (long)l * FDIM, ffib, TOKENS, FDIM, DMODEL, FDIM);
    gemm_wmma<0><<<gemmGrid(TOKENS, DMODEL), 256, 0, stream>>>(
        ffib, wf, bf_ + (long)l * DMODEL, tmpf, TOKENS, DMODEL, FDIM, DMODEL);
    residual_ln_kernel<<<TOKENS, 256, 0, stream>>>(
        h, hb, tmpf, ln2g + (long)l * DMODEL, ln2b + (long)l * DMODEL);
  }

  // classifier on last TITLE tokens
  {
    long cnt = (long)MCLS * DMODEL;
    gather_cls_kernel<<<dim3((unsigned)((cnt + 255) / 256)), 256, 0, stream>>>(hb, a320);
  }
  gemm_wmma<0><<<gemmGrid(MCLS, VPAD), 256, 0, stream>>>(
      a320, WcT, bcls, logits, MCLS, VPAD, DMODEL, VOCAB);

  nll_kernel<<<MCLS, 256, 0, stream>>>(logits, x, nllws);
  mean_kernel<<<1, 32, 0, stream>>>(nllws, out);
}
